// AugLag_55413668053294
// MI455X (gfx1250) — matrix operand order repaired and checked by reading, not compiled, not run on hardware
//
#include <hip/hip_runtime.h>

typedef __attribute__((ext_vector_type(16))) _Float16 v16h;
typedef __attribute__((ext_vector_type(8)))  _Float16 v8h;
typedef __attribute__((ext_vector_type(2)))  _Float16 v2h;
typedef __attribute__((ext_vector_type(8)))  float    v8f;
typedef __attribute__((ext_vector_type(4)))  float    v4f;

#define N_TOK_ 8192
#define OUT_F_ 4096
#define IN_F_  4096
#define STEP_  0.0078125f

// ---------------------------------------------------------------------------
// Kernel 1: reconstruct weight and downconvert to f16.
// w16[o*IN_F + i] = f16( (sum_j w_twos[o,i,j] * base[j]) * STEP )
// Pure streaming: ~570 MB @ 23.3 TB/s ~ 25 us.
// ---------------------------------------------------------------------------
__global__ void build_w16(const float* __restrict__ wt,
                          const float* __restrict__ base,
                          _Float16* __restrict__ w16) {
  size_t e = ((size_t)blockIdx.x * blockDim.x + threadIdx.x) * 2;
  float b[8];
#pragma unroll
  for (int j = 0; j < 8; ++j) b[j] = base[j];     // uniform -> scalar loads
  const v4f* p = (const v4f*)(wt + e * 8);
  float s[2];
#pragma unroll
  for (int q = 0; q < 2; ++q) {
    v4f a0 = p[q * 2 + 0];
    v4f a1 = p[q * 2 + 1];
    float acc = a0[0] * b[0] + a0[1] * b[1] + a0[2] * b[2] + a0[3] * b[3] +
                a1[0] * b[4] + a1[1] * b[5] + a1[2] * b[6] + a1[3] * b[7];
    s[q] = acc * STEP_;
  }
  v2h o;
  o[0] = (_Float16)s[0];
  o[1] = (_Float16)s[1];
  *(v2h*)(w16 + e) = o;
}

// ---------------------------------------------------------------------------
// Kernel 2: x (f32) -> x16 (f16), 8 elements per thread.
// ---------------------------------------------------------------------------
__global__ void conv_x16(const float* __restrict__ x, _Float16* __restrict__ x16) {
  size_t e = ((size_t)blockIdx.x * blockDim.x + threadIdx.x) * 8;
  v4f a = *(const v4f*)(x + e);
  v4f c = *(const v4f*)(x + e + 4);
  v8h o;
#pragma unroll
  for (int i = 0; i < 4; ++i) {
    o[i]     = (_Float16)a[i];
    o[4 + i] = (_Float16)c[i];
  }
  *(v8h*)(x16 + e) = o;
}

// ---------------------------------------------------------------------------
// Kernel 3: out = x16 @ w16^T + bias  via v_wmma_f32_16x16x32_f16.
// 256 threads = 8 wave32 (2x4 wave grid). Block tile 128M x 256N, wave tile
// 64x64 (4x4 fragments of 16x16 => 16 WMMA per 16 ds_load_b128, 1:1 ratio).
// BK=32, double-buffered LDS, padded row stride 40 halfs (80 B: 16B-aligned
// b128 LDS ops, conflict-free across 16 rows). Total LDS = 60 KB.
// Both A (x: MxK) and B (w: NxK) are K-contiguous -> plain b128 loads.
// ---------------------------------------------------------------------------
#define BM 128
#define BN 256
#define BK 32
#define LSTR 40                 // halfs per LDS row (32 + 8 pad)
#define KITERS (IN_F_ / BK)     // 128

__global__ __launch_bounds__(256)
void gemm_wmma(const _Float16* __restrict__ A,   // x16 [M,K]
               const _Float16* __restrict__ Bm,  // w16 [N,K]
               const float* __restrict__ bias,   // [N]
               float* __restrict__ C) {          // [M,N]
  __shared__ _Float16 sA[2][BM * LSTR];          // 2 x 10 KB
  __shared__ _Float16 sB[2][BN * LSTR];          // 2 x 20 KB

  const int tid  = threadIdx.x;
  const int lane = tid & 31;
  const int wave = tid >> 5;
  const int wm = wave >> 2;   // 0..1 : wave row (64 M each)
  const int wn = wave & 3;    // 0..3 : wave col (64 N each)
  const int lo = lane & 15;
  const int hi = lane >> 4;
  const size_t m0 = (size_t)blockIdx.y * BM;
  const size_t n0 = (size_t)blockIdx.x * BN;

  // Global->LDS staging map: 4 lanes per 32-half row, 64 rows per pass.
  const int lrow = tid >> 2;          // 0..63
  const int lcol = (tid & 3) * 8;     // 0,8,16,24

  v8f acc[4][4];
#pragma unroll
  for (int i = 0; i < 4; ++i)
#pragma unroll
    for (int j = 0; j < 4; ++j) acc[i][j] = {};

  v8h ra[2];   // A: 128 rows -> 2 passes
  v8h rb[4];   // B: 256 rows -> 4 passes

  // Prologue: stage tile 0.
#pragma unroll
  for (int p2 = 0; p2 < 2; ++p2)
    ra[p2] = *(const v8h*)(A + (m0 + lrow + p2 * 64) * IN_F_ + lcol);
#pragma unroll
  for (int p2 = 0; p2 < 4; ++p2)
    rb[p2] = *(const v8h*)(Bm + (n0 + lrow + p2 * 64) * IN_F_ + lcol);
#pragma unroll
  for (int p2 = 0; p2 < 2; ++p2)
    *(v8h*)&sA[0][(lrow + p2 * 64) * LSTR + lcol] = ra[p2];
#pragma unroll
  for (int p2 = 0; p2 < 4; ++p2)
    *(v8h*)&sB[0][(lrow + p2 * 64) * LSTR + lcol] = rb[p2];

  for (int it = 0; it < KITERS; ++it) {
    __syncthreads();
    const int cur = it & 1;

    // Prefetch next tile into registers (overlaps with WMMA below).
    if (it + 1 < KITERS) {
      const size_t kt = (size_t)(it + 1) * BK;
#pragma unroll
      for (int p2 = 0; p2 < 2; ++p2)
        ra[p2] = *(const v8h*)(A + (m0 + lrow + p2 * 64) * IN_F_ + kt + lcol);
#pragma unroll
      for (int p2 = 0; p2 < 4; ++p2)
        rb[p2] = *(const v8h*)(Bm + (n0 + lrow + p2 * 64) * IN_F_ + kt + lcol);
    }

    // Fragments from LDS per ISA layouts.
    v16h af[4];
    v16h bf[4];
#pragma unroll
    for (int mt = 0; mt < 4; ++mt) {
      // A 16-bit 16x32: lane(m=lo) elems 0..7 = K hi*8..+7, elems 8..15 = K 16+hi*8..
      const _Float16* pr = &sA[cur][(wm * 64 + mt * 16 + lo) * LSTR + hi * 8];
      v8h p0 = *(const v8h*)pr;
      v8h p1 = *(const v8h*)(pr + 16);
      af[mt] = __builtin_shufflevector(p0, p1, 0, 1, 2, 3, 4, 5, 6, 7,
                                       8, 9, 10, 11, 12, 13, 14, 15);
    }
#pragma unroll
    for (int nt = 0; nt < 4; ++nt) {
      // B 16-bit 32x16: lane(n=lo) elems 0..7 = K hi*8..+7, elems 8..15 = K 16+hi*8..
      const _Float16* pr = &sB[cur][(wn * 64 + nt * 16 + lo) * LSTR + hi * 8];
      v8h p0 = *(const v8h*)pr;
      v8h p1 = *(const v8h*)(pr + 16);
      bf[nt] = __builtin_shufflevector(p0, p1, 0, 1, 2, 3, 4, 5, 6, 7,
                                       8, 9, 10, 11, 12, 13, 14, 15);
    }

#pragma unroll
    for (int mt = 0; mt < 4; ++mt)
#pragma unroll
      for (int nt = 0; nt < 4; ++nt)
        acc[mt][nt] = __builtin_amdgcn_wmma_f32_16x16x32_f16(
            false, af[mt], false, bf[nt], (short)0, acc[mt][nt], false, false);

    // Stage next tile into the other buffer.
    if (it + 1 < KITERS) {
      const int nb = (it + 1) & 1;
#pragma unroll
      for (int p2 = 0; p2 < 2; ++p2)
        *(v8h*)&sA[nb][(lrow + p2 * 64) * LSTR + lcol] = ra[p2];
#pragma unroll
      for (int p2 = 0; p2 < 4; ++p2)
        *(v8h*)&sB[nb][(lrow + p2 * 64) * LSTR + lcol] = rb[p2];
    }
  }

  // Epilogue: C/D layout — lane n = lo, VGPR r holds row mt*16 + hi*8 + r.
#pragma unroll
  for (int nt = 0; nt < 4; ++nt) {
    const size_t col = n0 + wn * 64 + nt * 16 + lo;
    const float bv = bias[col];
#pragma unroll
    for (int mt = 0; mt < 4; ++mt) {
#pragma unroll
      for (int r2 = 0; r2 < 8; ++r2) {
        const size_t row = m0 + wm * 64 + mt * 16 + hi * 8 + r2;
        C[row * OUT_F_ + col] = acc[mt][nt][r2] + bv;
      }
    }
  }
}

// ---------------------------------------------------------------------------
extern "C" void kernel_launch(void* const* d_in, const int* in_sizes, int n_in,
                              void* d_out, int out_size, void* d_ws, size_t ws_size,
                              hipStream_t stream) {
  const float* x    = (const float*)d_in[0];  // [8192,4096]
  const float* wt   = (const float*)d_in[1];  // [4096,4096,8]
  const float* bias = (const float*)d_in[2];  // [4096]
  const float* base = (const float*)d_in[3];  // [8]
  float* out = (float*)d_out;                 // [8192,4096]

  _Float16* w16 = (_Float16*)d_ws;                                        // 32 MB
  _Float16* x16 = (_Float16*)((char*)d_ws + (size_t)OUT_F_ * IN_F_ * 2);  // 64 MB

  {
    const int nthr = (int)((size_t)OUT_F_ * IN_F_ / 2);   // 8,388,608
    build_w16<<<nthr / 256, 256, 0, stream>>>(wt, base, w16);
  }
  {
    const int nthr = (int)((size_t)N_TOK_ * IN_F_ / 8);   // 4,194,304
    conv_x16<<<nthr / 256, 256, 0, stream>>>(x, x16);
  }
  {
    dim3 grid(OUT_F_ / BN, N_TOK_ / BM);                  // 16 x 64
    gemm_wmma<<<grid, 256, 0, stream>>>(x16, w16, bias, out);
  }
}
